// MEGNN_33981781246507
// MI455X (gfx1250) — matrix operand order repaired and checked by reading, not compiled, not run on hardware
//
#include <hip/hip_runtime.h>
#include <stdint.h>

#define Gn   2
#define Ln   7
#define Hn   128
#define Nn   4096
#define En   258048
#define NINn 11

typedef __attribute__((ext_vector_type(16))) __bf16 v16bf;
typedef __attribute__((ext_vector_type(8)))  float  v8f;

union Frag8 { uint32_t u[8]; uint4 q[2]; v16bf v; };

__device__ __forceinline__ uint32_t pk_bf16(float a, float b) {
    uint32_t ua = __float_as_uint(a), ub = __float_as_uint(b);
    ua = (ua + 0x7FFFu + ((ua >> 16) & 1u)) >> 16;
    ub = (ub + 0x7FFFu + ((ub >> 16) & 1u)) >> 16;
    return ua | (ub << 16);
}
__device__ __forceinline__ unsigned short bf16_1(float a) {
    uint32_t ua = __float_as_uint(a);
    return (unsigned short)((ua + 0x7FFFu + ((ua >> 16) & 1u)) >> 16);
}
__device__ __forceinline__ float silu_f(float x) { return x / (1.f + __expf(-x)); }
__device__ __forceinline__ float sigm_f(float x) { return 1.f / (1.f + __expf(-x)); }

// ---- A fragment from LDS (16x32 bf16 tile, row-major, dword = K-pair) ------
__device__ __forceinline__ v16bf load_a_lds(const uint32_t* base, int strideDw,
                                            int mtile, int kt) {
    int lane = threadIdx.x & 31;
    const uint32_t* rp = base + (mtile * 16 + (lane & 15)) * strideDw;
    int pb = kt * 16 + ((lane >> 4) << 2);
    Frag8 f;
#pragma unroll
    for (int v = 0; v < 8; ++v) f.u[v] = rp[pb + v + (v & 4)];
    return f.v;
}
// ---- B fragment from global, fragment-ready layout [kt][nt][lane][8dw] -----
__device__ __forceinline__ v16bf load_b_frag(const uint32_t* wp, int kt, int nt) {
    int lane = threadIdx.x & 31;
    const uint32_t* p = wp + ((size_t)(kt * 8 + nt) * 256 + lane * 8);
    Frag8 f;
    f.q[0] = *(const uint4*)p;
    f.q[1] = *(const uint4*)(p + 4);
    return f.v;
}

// ---- weight packer: fp32 [mat][K][128] -> fragment-ready bf16 pair dwords --
// out[mat][kt][nt][lane][v], pair pb = kt*16 + (lane>>4)*4 + v + (v&4), n = nt*16+(lane&15)
__global__ void pack_kernel(const float* __restrict__ W, uint32_t* __restrict__ o,
                            int K, int KT) {
    int mat = blockIdx.y;
    const float* w = W + (size_t)mat * K * 128;
    uint32_t* op = o + (size_t)mat * KT * 2048;
    int idx = blockIdx.x * blockDim.x + threadIdx.x;
    if (idx >= KT * 2048) return;
    int kt = idx >> 11;
    int nt = (idx >> 8) & 7;
    int lane = (idx >> 3) & 31;
    int v = idx & 7;
    int pb = kt * 16 + ((lane >> 4) << 2) + v + (v & 4);
    int n = nt * 16 + (lane & 15);
    int k0 = 2 * pb, k1 = 2 * pb + 1;
    float a = (k0 < K) ? w[(size_t)k0 * 128 + n] : 0.f;
    float b = (k1 < K) ? w[(size_t)k1 * 128 + n] : 0.f;
    op[idx] = pk_bf16(a, b);
}

__global__ void embed_kernel(const float* __restrict__ h0, const float* __restrict__ W,
                             const float* __restrict__ b, float* __restrict__ h) {
    int idx = blockIdx.x * blockDim.x + threadIdx.x;   // over G*N*128
    int o = idx & 127, gn = idx >> 7;
    float s = b[o];
    const float* hp = h0 + (size_t)gn * NINn;
#pragma unroll
    for (int k = 0; k < NINn; ++k) s += hp[k] * W[k * 128 + o];
    h[idx] = s;
}

__global__ void radial_kernel(const float* __restrict__ x, const int* __restrict__ edges,
                              float* __restrict__ radial) {
    int idx = blockIdx.x * blockDim.x + threadIdx.x;   // over G*E
    if (idx >= Gn * En) return;
    int g = idx / En, e = idx - g * En;
    int r = edges[(size_t)g * 2 * En + e];
    int c = edges[(size_t)g * 2 * En + En + e];
    const float* xg = x + (size_t)g * Nn * 3;
    float dx = xg[r * 3 + 0] - xg[c * 3 + 0];
    float dy = xg[r * 3 + 1] - xg[c * 3 + 1];
    float dz = xg[r * 3 + 2] - xg[c * 3 + 2];
    radial[idx] = dx * dx + dy * dy + dz * dz;
}

__global__ void zero_kernel(float* __restrict__ p) {
    p[blockIdx.x * blockDim.x + threadIdx.x] = 0.f;
}

// ---------------- fused edge kernel: gather -> MLP(WMMA) -> gate -> scatter ---
#define AS_DW 146   // 292 bf16 per row
#define M1_DW 66    // 132 bf16 per row
#define MS_ST 130   // fp32 stride (overlays As region)

__global__ __launch_bounds__(128) void edge_kernel(
    const float* __restrict__ h, float* __restrict__ agg,
    const float* __restrict__ radial, const float* __restrict__ eattr,
    const float* __restrict__ emask, const int* __restrict__ edges,
    const uint32_t* __restrict__ eW1p, const uint32_t* __restrict__ eW2p,
    const float* __restrict__ e_b1, const float* __restrict__ e_b2,
    const float* __restrict__ att_W, const float* __restrict__ att_b, int layer) {
    __shared__ uint32_t smem[64 * AS_DW + 64 * M1_DW];
    __shared__ float sdot[128];
    __shared__ float gate[64];
    __shared__ int rowS[64];

    int g = blockIdx.y, t = threadIdx.x;
    int e0 = blockIdx.x * 64;
    int gl = g * Ln + layer;
    const float* hg = h + (size_t)g * Nn * Hn;
    const int* rowp = edges + (size_t)g * 2 * En;
    const int* colp = rowp + En;

    // ---- build A tile [64][292] bf16: h[row] | h[col] | radial | eattr | 0-pad
    // 4 rows per iteration; 32 lanes x float4 cover a 128-float row.
    {
        int tr = t >> 5;            // row within group of 4
        int tc = (t & 31) << 2;     // starting column (float)
        for (int r4 = 0; r4 < 16; ++r4) {
            int r = r4 * 4 + tr;
            int e = e0 + r;
            int ri = rowp[e], ci = colp[e];
            float4 a = *(const float4*)&hg[(size_t)ri * Hn + tc];
            float4 b = *(const float4*)&hg[(size_t)ci * Hn + tc];
            uint32_t* dst = smem + r * AS_DW;
            uint2 pa; pa.x = pk_bf16(a.x, a.y); pa.y = pk_bf16(a.z, a.w);
            uint2 pb; pb.x = pk_bf16(b.x, b.y); pb.y = pk_bf16(b.z, b.w);
            *(uint2*)&dst[tc >> 1]      = pa;
            *(uint2*)&dst[64 + (tc >> 1)] = pb;
        }
    }
    {
        unsigned short* Ash = (unsigned short*)smem;
        for (int idx = t; idx < 64 * 36; idx += 128) {
            int r = idx / 36, c = idx - r * 36;
            int e = e0 + r;
            float v = 0.f;
            if (c == 0) v = radial[(size_t)g * En + e];
            else if (c == 1) v = eattr[((size_t)g * En + e) * 2 + 0];
            else if (c == 2) v = eattr[((size_t)g * En + e) * 2 + 1];
            Ash[r * 292 + 256 + c] = bf16_1(v);
        }
    }
    if (t < 64) rowS[t] = rowp[e0 + t];
    __syncthreads();

    int wave = t >> 5, lane = t & 31;
    int nl = lane & 15;
    int mrow = wave * 16 + ((lane >> 4) << 3);

    // ---- GEMM1: [64x288] x [288x128] bf16 -> silu -> M1 (bf16 in LDS)
    const uint32_t* W1 = eW1p + (size_t)gl * 9 * 2048;
    const float* b1 = e_b1 + (size_t)gl * Hn;
    v8f acc[8] = {};
    for (int kt = 0; kt < 9; ++kt) {
        v16bf a = load_a_lds(smem, AS_DW, wave, kt);
#pragma unroll
        for (int nt = 0; nt < 8; ++nt) {
            v16bf b = load_b_frag(W1, kt, nt);
            acc[nt] = __builtin_amdgcn_wmma_f32_16x16x32_bf16(
                false, a, false, b, (short)0, acc[nt], false, false);
        }
    }
    unsigned short* M1h = (unsigned short*)(smem + 64 * AS_DW);
#pragma unroll
    for (int nt = 0; nt < 8; ++nt)
#pragma unroll
        for (int i = 0; i < 8; ++i) {
            float v = silu_f(acc[nt][i] + b1[nt * 16 + nl]);
            M1h[(mrow + i) * 132 + nt * 16 + nl] = bf16_1(v);
        }

    // ---- GEMM2: [64x128] x [128x128]
    const uint32_t* W2 = eW2p + (size_t)gl * 4 * 2048;
    const float* b2 = e_b2 + (size_t)gl * Hn;
    v8f acc2[8] = {};
    for (int kt = 0; kt < 4; ++kt) {
        v16bf a = load_a_lds(smem + 64 * AS_DW, M1_DW, wave, kt);
#pragma unroll
        for (int nt = 0; nt < 8; ++nt) {
            v16bf b = load_b_frag(W2, kt, nt);
            acc2[nt] = __builtin_amdgcn_wmma_f32_16x16x32_bf16(
                false, a, false, b, (short)0, acc2[nt], false, false);
        }
    }
    __syncthreads();                       // all waves done reading As before overlay
    float* Ms = (float*)smem;              // fp32 m2 tile, stride MS_ST
#pragma unroll
    for (int nt = 0; nt < 8; ++nt)
#pragma unroll
        for (int i = 0; i < 8; ++i)
            Ms[(mrow + i) * MS_ST + nt * 16 + nl] = silu_f(acc2[nt][i] + b2[nt * 16 + nl]);
    __syncthreads();

    // ---- attention gate: sigmoid(m . att_W + b) * edge_mask
    {
        int m = t >> 1, half = t & 1;
        const float* aw = att_W + (size_t)gl * Hn;
        float p = 0.f;
        for (int n = half * 64; n < half * 64 + 64; ++n) p += Ms[m * MS_ST + n] * aw[n];
        sdot[t] = p;
    }
    __syncthreads();
    if (t < 64)
        gate[t] = sigm_f(sdot[2 * t] + sdot[2 * t + 1] + att_b[gl]) *
                  emask[(size_t)g * En + e0 + t];
    __syncthreads();

    // ---- scatter-add into agg (segment_sum over row)
    float* ag = agg + (size_t)g * Nn * Hn;
    for (int m = 0; m < 64; ++m)
        atomicAdd(&ag[(size_t)rowS[m] * Hn + t], Ms[m * MS_ST + t] * gate[m]);
}

// ---------------- fused node kernel: [h|agg|h0] -> MLP(WMMA) -> residual -----
__global__ __launch_bounds__(128) void node_kernel(
    const float* __restrict__ h0, float* __restrict__ h, const float* __restrict__ agg,
    const uint32_t* __restrict__ nW1p, const uint32_t* __restrict__ nW2p,
    const float* __restrict__ n_b1, const float* __restrict__ n_b2, int layer) {
    __shared__ uint32_t smem[64 * AS_DW + 64 * M1_DW];
    int g = blockIdx.y, t = threadIdx.x;
    int n0 = blockIdx.x * 64;
    int gl = g * Ln + layer;
    const float* hg = h + (size_t)g * Nn * Hn;
    const float* ag = agg + (size_t)g * Nn * Hn;

    {
        int tr = t >> 5;
        int tc = (t & 31) << 2;
        for (int r4 = 0; r4 < 16; ++r4) {
            int r = r4 * 4 + tr;
            int node = n0 + r;
            float4 a = *(const float4*)&hg[(size_t)node * Hn + tc];
            float4 b = *(const float4*)&ag[(size_t)node * Hn + tc];
            uint32_t* dst = smem + r * AS_DW;
            uint2 pa; pa.x = pk_bf16(a.x, a.y); pa.y = pk_bf16(a.z, a.w);
            uint2 pb; pb.x = pk_bf16(b.x, b.y); pb.y = pk_bf16(b.z, b.w);
            *(uint2*)&dst[tc >> 1]        = pa;
            *(uint2*)&dst[64 + (tc >> 1)] = pb;
        }
    }
    {
        unsigned short* Ash = (unsigned short*)smem;
        for (int idx = t; idx < 64 * 36; idx += 128) {
            int r = idx / 36, c = idx - r * 36;
            float v = (c < NINn) ? h0[((size_t)g * Nn + n0 + r) * NINn + c] : 0.f;
            Ash[r * 292 + 256 + c] = bf16_1(v);
        }
    }
    __syncthreads();

    int wave = t >> 5, lane = t & 31;
    int nl = lane & 15;
    int mrow = wave * 16 + ((lane >> 4) << 3);

    const uint32_t* W1 = nW1p + (size_t)gl * 9 * 2048;
    const float* b1 = n_b1 + (size_t)gl * Hn;
    v8f acc[8] = {};
    for (int kt = 0; kt < 9; ++kt) {
        v16bf a = load_a_lds(smem, AS_DW, wave, kt);
#pragma unroll
        for (int nt = 0; nt < 8; ++nt) {
            v16bf b = load_b_frag(W1, kt, nt);
            acc[nt] = __builtin_amdgcn_wmma_f32_16x16x32_bf16(
                false, a, false, b, (short)0, acc[nt], false, false);
        }
    }
    unsigned short* M1h = (unsigned short*)(smem + 64 * AS_DW);
#pragma unroll
    for (int nt = 0; nt < 8; ++nt)
#pragma unroll
        for (int i = 0; i < 8; ++i) {
            float v = silu_f(acc[nt][i] + b1[nt * 16 + nl]);
            M1h[(mrow + i) * 132 + nt * 16 + nl] = bf16_1(v);
        }

    const uint32_t* W2 = nW2p + (size_t)gl * 4 * 2048;
    const float* b2 = n_b2 + (size_t)gl * Hn;
    v8f acc2[8] = {};
    for (int kt = 0; kt < 4; ++kt) {
        v16bf a = load_a_lds(smem + 64 * AS_DW, M1_DW, wave, kt);
#pragma unroll
        for (int nt = 0; nt < 8; ++nt) {
            v16bf b = load_b_frag(W2, kt, nt);
            acc2[nt] = __builtin_amdgcn_wmma_f32_16x16x32_bf16(
                false, a, false, b, (short)0, acc2[nt], false, false);
        }
    }
    // residual: h += upd  (each (node,col) owned by exactly one thread)
    float* hw = h + (size_t)g * Nn * Hn;
#pragma unroll
    for (int nt = 0; nt < 8; ++nt)
#pragma unroll
        for (int i = 0; i < 8; ++i) {
            size_t idx = (size_t)(n0 + mrow + i) * Hn + nt * 16 + nl;
            hw[idx] = hw[idx] + acc2[nt][i] + b2[nt * 16 + nl];
        }
}

// ---------------- node decoder + masked per-molecule pooling -----------------
__global__ __launch_bounds__(128) void node_dec_kernel(
    const float* __restrict__ h, const float* __restrict__ nmask,
    const float* __restrict__ W1, const float* __restrict__ b1,
    const float* __restrict__ W2, const float* __restrict__ b2,
    float* __restrict__ pooled) {
    __shared__ float tt[64 * 128];
    int g = blockIdx.y, mol = blockIdx.x, o = threadIdx.x;
    const float* hg = h + ((size_t)g * Nn + mol * 64) * Hn;
    const float* w1 = W1 + (size_t)g * Hn * Hn;
    const float* w2 = W2 + (size_t)g * Hn * Hn;
    float bb1 = b1[g * Hn + o], bb2 = b2[g * Hn + o];
    for (int node = 0; node < 64; ++node) {
        float s = bb1;
        for (int k = 0; k < 128; ++k) s += hg[node * 128 + k] * w1[k * 128 + o];
        tt[node * 128 + o] = silu_f(s);
    }
    __syncthreads();
    float pool = 0.f;
    for (int node = 0; node < 64; ++node) {
        float s = bb2;
        for (int k = 0; k < 128; ++k) s += tt[node * 128 + k] * w2[k * 128 + o];
        pool += s * nmask[(size_t)g * Nn + mol * 64 + node];
    }
    pooled[((size_t)g * 64 + mol) * Hn + o] = pool;
}

__global__ __launch_bounds__(256) void graph_dec_kernel(
    const float* __restrict__ pooled, const float* __restrict__ W1,
    const float* __restrict__ b1, const float* __restrict__ W2,
    const float* __restrict__ b2, float* __restrict__ out) {
    __shared__ float comb[256];
    __shared__ float red[256];
    int t = threadIdx.x;
    for (int b = 0; b < 64; ++b) {
        comb[t] = pooled[((size_t)(t >> 7) * 64 + b) * Hn + (t & 127)];
        __syncthreads();
        float s = b1[t];
        for (int k = 0; k < 256; ++k) s += comb[k] * W1[k * 256 + t];
        red[t] = silu_f(s) * W2[t];
        __syncthreads();
        for (int off = 128; off; off >>= 1) {
            if (t < off) red[t] += red[t + off];
            __syncthreads();
        }
        if (t == 0) out[b] = red[0] + b2[0];
        __syncthreads();
    }
}

extern "C" void kernel_launch(void* const* d_in, const int* in_sizes, int n_in,
                              void* d_out, int out_size, void* d_ws, size_t ws_size,
                              hipStream_t stream) {
    const float* h0    = (const float*)d_in[0];
    const float* x     = (const float*)d_in[1];
    const float* eattr = (const float*)d_in[2];
    const float* nmask = (const float*)d_in[3];
    const float* emask = (const float*)d_in[4];
    const float* emb_W = (const float*)d_in[5];
    const float* emb_b = (const float*)d_in[6];
    const float* e_W1  = (const float*)d_in[7];
    const float* e_b1  = (const float*)d_in[8];
    const float* e_W2  = (const float*)d_in[9];
    const float* e_b2  = (const float*)d_in[10];
    const float* att_W = (const float*)d_in[11];
    const float* att_b = (const float*)d_in[12];
    const float* n_W1  = (const float*)d_in[13];
    const float* n_b1  = (const float*)d_in[14];
    const float* n_W2  = (const float*)d_in[15];
    const float* n_b2  = (const float*)d_in[16];
    const float* nd_W1 = (const float*)d_in[17];
    const float* nd_b1 = (const float*)d_in[18];
    const float* nd_W2 = (const float*)d_in[19];
    const float* nd_b2 = (const float*)d_in[20];
    const float* gd_W1 = (const float*)d_in[21];
    const float* gd_b1 = (const float*)d_in[22];
    const float* gd_W2 = (const float*)d_in[23];
    const float* gd_b2 = (const float*)d_in[24];
    const int*   edges = (const int*)d_in[25];
    float* out = (float*)d_out;

    char* ws = (char*)d_ws;
    size_t off = 0;
    float* h      = (float*)(ws + off); off += (size_t)Gn * Nn * Hn * 4;
    float* agg    = (float*)(ws + off); off += (size_t)Gn * Nn * Hn * 4;
    float* radial = (float*)(ws + off); off += (size_t)Gn * En * 4;
    float* pooled = (float*)(ws + off); off += (size_t)Gn * 64 * Hn * 4;
    uint32_t* eW1p = (uint32_t*)(ws + off); off += (size_t)Gn * Ln * 9 * 2048 * 4;
    uint32_t* eW2p = (uint32_t*)(ws + off); off += (size_t)Gn * Ln * 4 * 2048 * 4;
    uint32_t* nW1p = (uint32_t*)(ws + off); off += (size_t)Gn * Ln * 9 * 2048 * 4;
    uint32_t* nW2p = (uint32_t*)(ws + off); off += (size_t)Gn * Ln * 4 * 2048 * 4;

    pack_kernel<<<dim3(72, Gn * Ln), 256, 0, stream>>>(e_W1, eW1p, 259, 9);
    pack_kernel<<<dim3(32, Gn * Ln), 256, 0, stream>>>(e_W2, eW2p, 128, 4);
    pack_kernel<<<dim3(72, Gn * Ln), 256, 0, stream>>>(n_W1, nW1p, 267, 9);
    pack_kernel<<<dim3(32, Gn * Ln), 256, 0, stream>>>(n_W2, nW2p, 128, 4);

    embed_kernel<<<(Gn * Nn * Hn) / 256, 256, 0, stream>>>(h0, emb_W, emb_b, h);
    radial_kernel<<<(Gn * En + 255) / 256, 256, 0, stream>>>(x, edges, radial);

    for (int l = 0; l < Ln; ++l) {
        zero_kernel<<<(Gn * Nn * Hn) / 256, 256, 0, stream>>>(agg);
        edge_kernel<<<dim3(En / 64, Gn), 128, 0, stream>>>(
            h, agg, radial, eattr, emask, edges, eW1p, eW2p, e_b1, e_b2,
            att_W, att_b, l);
        node_kernel<<<dim3(Nn / 64, Gn), 128, 0, stream>>>(
            h0, h, agg, nW1p, nW2p, n_b1, n_b2, l);
    }
    node_dec_kernel<<<dim3(64, Gn), 128, 0, stream>>>(
        h, nmask, nd_W1, nd_b1, nd_W2, nd_b2, pooled);
    graph_dec_kernel<<<1, 256, 0, stream>>>(pooled, gd_W1, gd_b1, gd_W2, gd_b2, out);
}
